// RingLoss_31284541784142
// MI455X (gfx1250) — matrix operand order repaired and checked by reading, not compiled, not run on hardware
//
#include <hip/hip_runtime.h>
#include <hip/hip_bf16.h>

typedef __attribute__((ext_vector_type(16))) _Float16 v16h;
typedef __attribute__((ext_vector_type(8)))  _Float16 v8h;
typedef __attribute__((ext_vector_type(8)))  float    v8f;

#define N_ROWS 4096
#define DIM    512
#define M2     8192          // 2N rows of Z = [z1; z2]
#define THRK   409           // int(4096 * 0.1)
#define BLK    128           // C tile is BLK x BLK
#define BK     32            // K tile (matches wmma f16 K=32)
#define LDSS   40            // LDS row stride in halfs (32 + 8 pad -> bank-conflict-free b128 reads)

// ---------------- CDNA5 async global->LDS copy (ASYNCcnt path) ----------------

__device__ __forceinline__ void async_b128(unsigned lds_off, const void* gaddr) {
    // GLOBAL_LOAD_ASYNC_TO_LDS_B128: vdst = per-lane LDS byte address,
    // vaddr = per-lane 64-bit global address, saddr = off. Tracked by ASYNCcnt.
    asm volatile("global_load_async_to_lds_b128 %0, %1, off"
                 :
                 : "v"(lds_off), "v"(gaddr)
                 : "memory");
}

__device__ __forceinline__ void wait_async_all() {
    asm volatile("s_wait_asynccnt 0x0" ::: "memory");
}

// low 32 bits of a flat pointer into __shared__ == LDS byte address (aperture rule, ISA 10.2)
__device__ __forceinline__ unsigned lds_addr(const void* p) {
    return (unsigned)(unsigned long long)p;
}

// ---------------- helpers ----------------

__device__ __forceinline__ float block_reduce(float v, float* red) {
    int tid = threadIdx.x;
    red[tid] = v;
    __syncthreads();
    for (int s = 128; s > 0; s >>= 1) {
        if (tid < s) red[tid] += red[tid + s];
        __syncthreads();
    }
    float r = red[0];
    __syncthreads();
    return r;
}

// monotone float <-> uint mapping (ascending order preserved)
__device__ __forceinline__ unsigned f2ord(float f) {
    unsigned u = __float_as_uint(f);
    return (u & 0x80000000u) ? ~u : (u | 0x80000000u);
}
__device__ __forceinline__ float ord2f(unsigned o) {
    unsigned u = (o & 0x80000000u) ? (o & 0x7fffffffu) : ~o;
    return __uint_as_float(u);
}

// exact k-th largest over M2 values cached in LDS, via 4x 8-bit radix passes
__device__ unsigned kth_largest_key(const float* rowv, unsigned* hist,
                                    int* sel_s, int* k_s, int k) {
    const int tid = threadIdx.x;
    unsigned prefix = 0, mask = 0;
    if (tid == 0) *k_s = k;
    __syncthreads();
    for (int byte = 3; byte >= 0; --byte) {
        const int shift = byte * 8;
        hist[tid] = 0;
        __syncthreads();
        for (int i = tid; i < M2; i += 256) {
            unsigned key = f2ord(rowv[i]);
            if ((key & mask) == prefix)
                atomicAdd(&hist[(key >> shift) & 255u], 1u);
        }
        __syncthreads();
        if (tid == 0) {
            int kk = *k_s, cum = 0, sel = 0;
            for (int b = 255; b >= 0; --b) {
                int c = (int)hist[b];
                if (cum + c >= kk) { sel = b; *k_s = kk - cum; break; }
                cum += c;
            }
            *sel_s = sel;
        }
        __syncthreads();
        prefix |= ((unsigned)(*sel_s)) << shift;
        mask   |= 0xFFu << shift;
        __syncthreads();
    }
    return prefix;
}

// ---------------- kernel 1: L2-normalize rows, emit f16 Z ----------------

__global__ __launch_bounds__(256) void normalize_rows(const float* __restrict__ h1,
                                                      const float* __restrict__ h2,
                                                      _Float16* __restrict__ Zh) {
    __shared__ float red[256];
    const int r   = blockIdx.x;
    const int tid = threadIdx.x;
    const float* src = (r < N_ROWS) ? (h1 + (size_t)r * DIM)
                                    : (h2 + (size_t)(r - N_ROWS) * DIM);
    float v0 = src[tid];
    float v1 = src[tid + 256];
    float ss = block_reduce(v0 * v0 + v1 * v1, red);
    float scale = 1.0f / fmaxf(sqrtf(ss), 1e-12f);
    _Float16* dst = Zh + (size_t)r * DIM;
    dst[tid]       = (_Float16)(v0 * scale);
    dst[tid + 256] = (_Float16)(v1 * scale);
}

// ---------------- kernel 2: S = Z * Z^T via v_wmma_f32_16x16x32_f16 ----------------
// Block: 256 threads = 8 waves (wave32). Each block computes a 128x128 tile of S.
// Wave (wy in 0..3, wx in 0..1) owns a 32x64 sub-tile = 2x4 grid of 16x16 WMMA tiles.
// Tile staging uses CDNA5 async global->LDS copies, double-buffered:
//   iter kk: issue async(tile kk+1 -> buf^1) | compute(buf) | s_wait_asynccnt 0 | barrier

__global__ __launch_bounds__(256) void gram_wmma(const _Float16* __restrict__ Zh,
                                                 float* __restrict__ S) {
    __shared__ _Float16 sA[2][BLK * LDSS];
    __shared__ _Float16 sB[2][BLK * LDSS];

    const int tid   = threadIdx.x;
    const int lane  = tid & 31;          // wave32
    const int w     = tid >> 5;
    const int waveM = (w >> 1) * 32;
    const int waveN = (w & 1) * 64;

    const int m0 = blockIdx.y * BLK;
    const int n0 = blockIdx.x * BLK;

    // cooperative tile loader: thread t -> row t/2, 16-half segment (t&1)
    const int ldRow = tid >> 1;
    const int ldSeg = (tid & 1) * 16;

    const _Float16* gA = Zh + (size_t)(m0 + ldRow) * DIM + ldSeg;
    const _Float16* gB = Zh + (size_t)(n0 + ldRow) * DIM + ldSeg;

    unsigned ldsAo[2], ldsBo[2];
    ldsAo[0] = lds_addr(&sA[0][ldRow * LDSS + ldSeg]);
    ldsAo[1] = lds_addr(&sA[1][ldRow * LDSS + ldSeg]);
    ldsBo[0] = lds_addr(&sB[0][ldRow * LDSS + ldSeg]);
    ldsBo[1] = lds_addr(&sB[1][ldRow * LDSS + ldSeg]);

    v8f zero = {};
    v8f acc[2][4];
#pragma unroll
    for (int i = 0; i < 2; ++i)
#pragma unroll
        for (int j = 0; j < 4; ++j) acc[i][j] = zero;

    // prologue: async-copy K-tile 0 into buffer 0
    async_b128(ldsAo[0],      gA);
    async_b128(ldsAo[0] + 16, gA + 8);
    async_b128(ldsBo[0],      gB);
    async_b128(ldsBo[0] + 16, gB + 8);
    wait_async_all();
    __syncthreads();

    const int lrow = lane & 15;
    const int kloA = (lane < 16) ? 0 : 8;    // A layout: K {0..7,16..23} | {8..15,24..31}
    const int kbB  = (lane < 16) ? 0 : 8;    // B layout: K {0..7,16..23} | {8..15,24..31}

    for (int kk = 0; kk < DIM / BK; ++kk) {
        const int buf = kk & 1;

        if (kk + 1 < DIM / BK) {             // overlap next tile copy with compute
            const int nb = buf ^ 1;
            gA += BK; gB += BK;
            async_b128(ldsAo[nb],      gA);
            async_b128(ldsAo[nb] + 16, gA + 8);
            async_b128(ldsBo[nb],      gB);
            async_b128(ldsBo[nb] + 16, gB + 8);
        }

        v16h af[2], bf[4];
#pragma unroll
        for (int tm = 0; tm < 2; ++tm) {
            const _Float16* p = &sA[buf][(waveM + tm * 16 + lrow) * LDSS + kloA];
            v8h lo = *(const v8h*)(p);        // halfs [klo, klo+8)
            v8h hi = *(const v8h*)(p + 16);   // halfs [klo+16, klo+24)
            af[tm] = __builtin_shufflevector(lo, hi,
                     0,1,2,3,4,5,6,7,8,9,10,11,12,13,14,15);
        }
#pragma unroll
        for (int tn = 0; tn < 4; ++tn) {
            const _Float16* p = &sB[buf][(waveN + tn * 16 + lrow) * LDSS + kbB];
            v8h b0 = *(const v8h*)(p);        // halfs [kb, kb+8)
            v8h b1 = *(const v8h*)(p + 16);   // halfs [kb+16, kb+24)
            bf[tn] = __builtin_shufflevector(b0, b1,
                     0,1,2,3,4,5,6,7,8,9,10,11,12,13,14,15);
        }
#pragma unroll
        for (int tm = 0; tm < 2; ++tm)
#pragma unroll
            for (int tn = 0; tn < 4; ++tn)
                acc[tm][tn] = __builtin_amdgcn_wmma_f32_16x16x32_f16(
                    false, af[tm], false, bf[tn], (short)0, acc[tm][tn],
                    false, false);

        wait_async_all();     // next tile landed in LDS (overlapped with WMMAs above)
        __syncthreads();      // all waves done reading buf + all copies visible
    }

    // C layout: VGPR r, lanes<16 -> M=r ; lanes>=16 -> M=8+r ; N = lane&15
    const int cRowOff = (lane >> 4) * 8;
    const int cCol    = lane & 15;
#pragma unroll
    for (int tm = 0; tm < 2; ++tm)
#pragma unroll
        for (int tn = 0; tn < 4; ++tn) {
            size_t base = (size_t)(m0 + waveM + tm * 16 + cRowOff) * M2
                        + (size_t)(n0 + waveN + tn * 16 + cCol);
#pragma unroll
            for (int r2 = 0; r2 < 8; ++r2)
                S[base + (size_t)r2 * M2] = acc[tm][tn][r2];
        }
}

// ---------------- kernel 3: per-row trimmed exp-sum (exact, sort-free) ----------------

__global__ __launch_bounds__(256) void rowstat(const float* __restrict__ S,
                                               float* __restrict__ logneg,
                                               float* __restrict__ cosv) {
    __shared__ float    rowv[M2];     // 32 KB row cache
    __shared__ unsigned hist[256];
    __shared__ float    red[256];
    __shared__ int      sel_s, k_s;

    const int tid = threadIdx.x;
    const int r   = blockIdx.x;
    const float* srow = S + (size_t)r * M2;

    float tpart = 0.0f;
    for (int i = tid; i < M2; i += 256) {
        float v = srow[i];
        rowv[i] = v;
        tpart += expf(2.0f * v);      // exp(v / tau), tau = 0.5
    }
    float total = block_reduce(tpart, red);   // internal barriers also publish rowv

    // 409th largest, and 409th smallest (= (M2-409+1)-th largest)
    unsigned keyHi = kth_largest_key(rowv, hist, &sel_s, &k_s, THRK);
    unsigned keyLo = kth_largest_key(rowv, hist, &sel_s, &k_s, M2 - THRK + 1);
    float Thi = ord2f(keyHi);
    float Tlo = ord2f(keyLo);

    float sg = 0.f, sl = 0.f, cg = 0.f, cl = 0.f;
    for (int i = tid; i < M2; i += 256) {
        float v = rowv[i];
        if (v > Thi) { sg += expf(2.0f * v); cg += 1.0f; }
        if (v < Tlo) { sl += expf(2.0f * v); cl += 1.0f; }
    }
    sg = block_reduce(sg, red);
    sl = block_reduce(sl, red);
    cg = block_reduce(cg, red);
    cl = block_reduce(cl, red);

    if (tid == 0) {
        // tie-exact trimmed sum: strictly-outside values + duplicated boundary values
        float topSum = sg + ((float)THRK - cg) * expf(2.0f * Thi);
        float botSum = sl + ((float)THRK - cl) * expf(2.0f * Tlo);
        float trimmed = total - topSum - botSum;
        logneg[r] = logf(trimmed);
        if (r < N_ROWS) cosv[r] = rowv[N_ROWS + r];  // z1_i . z2_i
    }
}

// ---------------- kernel 4: final mean ----------------

__global__ __launch_bounds__(256) void finalize(const float* __restrict__ logneg,
                                                const float* __restrict__ cosv,
                                                float* __restrict__ out) {
    __shared__ float red[256];
    const int tid = threadIdx.x;
    float p = 0.f;
    for (int i = tid; i < N_ROWS; i += 256)
        p += 0.5f * (logneg[i] + logneg[N_ROWS + i]) - 2.0f * cosv[i];
    float t = block_reduce(p, red);
    if (tid == 0) out[0] = t / (float)N_ROWS;
}

// ---------------- launcher ----------------

extern "C" void kernel_launch(void* const* d_in, const int* in_sizes, int n_in,
                              void* d_out, int out_size, void* d_ws, size_t ws_size,
                              hipStream_t stream) {
    (void)in_sizes; (void)n_in; (void)out_size; (void)ws_size;
    const float* h1 = (const float*)d_in[0];
    const float* h2 = (const float*)d_in[1];

    // workspace layout: Zh (f16, 8 MB) | S (f32, 256 MB) | logneg (8192 f32) | cosv (4096 f32)
    char* ws = (char*)d_ws;
    _Float16* Zh   = (_Float16*)ws;
    float*    S    = (float*)(ws + (size_t)M2 * DIM * sizeof(_Float16));
    float* logneg  = (float*)((char*)S + (size_t)M2 * M2 * sizeof(float));
    float* cosv    = logneg + M2;
    float* out     = (float*)d_out;

    normalize_rows<<<M2, 256, 0, stream>>>(h1, h2, Zh);
    gram_wmma<<<dim3(M2 / BLK, M2 / BLK), 256, 0, stream>>>(Zh, S);
    rowstat<<<M2, 256, 0, stream>>>(S, logneg, cosv);
    finalize<<<1, 256, 0, stream>>>(logneg, cosv, out);
}
